// DeepSeekMoE_91122026152104
// MI455X (gfx1250) — compile-verified, matrix operand order checked by reading, not run on hardware
//
#include <hip/hip_runtime.h>
#include <hip/hip_bf16.h>

// ---------------------------------------------------------------------------
// DeepSeek-style MoE for MI455X (gfx1250, wave32, WMMA bf16 16x16x32)
//   k0: f32->bf16 weight conversion
//   k1: fused RMSNorm + sigmoid router + top-2 gather-list build
//   k2: shared expert  (dense, 32-token tiles, async LDS staging, WMMA)
//   k3: routed experts (gathered, gate-scaled atomic scatter, WMMA)
// ---------------------------------------------------------------------------

#define MOE_USE_ASYNC_LDS 1   // set 0 to fall back to VGPR-staged LDS copy

typedef __bf16 bf16_t;
typedef bf16_t v16bf __attribute__((ext_vector_type(16)));
typedef bf16_t v8bf  __attribute__((ext_vector_type(8)));
typedef float  v8f   __attribute__((ext_vector_type(8)));

#define B_      4
#define T_      1024
#define N_TOK   (B_ * T_)   // 4096
#define D_      1024
#define F_      512
#define E_      8
#define MT_     32          // token rows per workgroup tile

#define GATE_MAX_  30.0f
#define LIN_MIN_  -100.0f
#define LIN_MAX_   100.0f
#define EPS_RMS_   1e-6f

// Padded LDS strides (elements) so 16 fragment lanes hit disjoint bank groups
#define HS_STRIDE  (D_ + 8)   // 1032 elems -> 2064 B row stride
#define ACT_STRIDE (F_ + 8)   // 520 elems  -> 1040 B row stride

#define HS_BYTES   (MT_ * HS_STRIDE * 2)   // 66048
#define ACT_BYTES  (MT_ * ACT_STRIDE * 2)  // 33280
#define SMEM_TOTAL (HS_BYTES + ACT_BYTES + MT_ * 4 + MT_ * 4)  // ~99.8 KB

__device__ __forceinline__ bf16_t f2bf(float f) {
  unsigned u = __builtin_bit_cast(unsigned, f);
  unsigned r = (u + 0x7FFFu + ((u >> 16) & 1u)) >> 16;  // round-to-nearest-even
  unsigned short s = (unsigned short)r;
  return __builtin_bit_cast(bf16_t, s);
}

// Load one 16x32 bf16 WMMA operand fragment for this lane.
// Layout (ISA 7.12.2, 16-bit A 16x32): lane half 0 holds K = k0+[0..7] and
// k0+16+[0..7]; lane half 1 holds K = k0+8+[0..7] and k0+24+[0..7].
__device__ __forceinline__ v16bf frag_from(const bf16_t* rowbase, int k0, int half) {
  const v8bf* p0 = reinterpret_cast<const v8bf*>(rowbase + k0 + half * 8);
  const v8bf* p1 = reinterpret_cast<const v8bf*>(rowbase + k0 + 16 + half * 8);
  v8bf a = *p0;
  v8bf b = *p1;
  v16bf r;
#pragma unroll
  for (int i = 0; i < 8; ++i) { r[i] = a[i]; r[i + 8] = b[i]; }
  return r;
}

// Per-lane async DMA: 16 bytes global -> LDS, tracked on ASYNCcnt.
__device__ __forceinline__ void async_g2l_b128(void* lds_ptr, const void* gptr) {
#if MOE_USE_ASYNC_LDS
  unsigned lds_addr = (unsigned)(unsigned long long)lds_ptr;  // LDS offset = addr[31:0]
  unsigned long long ga = (unsigned long long)gptr;
  asm volatile("global_load_async_to_lds_b128 %0, %1, off"
               :: "v"(lds_addr), "v"(ga) : "memory");
#else
  *reinterpret_cast<v8bf*>(lds_ptr) = *reinterpret_cast<const v8bf*>(gptr);
#endif
}

__device__ __forceinline__ void wait_async_then_none() {
#if MOE_USE_ASYNC_LDS
  asm volatile("s_wait_asynccnt 0x0" ::: "memory");
#endif
}

// ---------------------------------------------------------------------------
// Kernel 0: f32 -> bf16 weight conversion
// ---------------------------------------------------------------------------
__global__ __launch_bounds__(256)
void cvt_bf16_kernel(const float* __restrict__ src, bf16_t* __restrict__ dst, int n) {
  int i = blockIdx.x * blockDim.x + threadIdx.x;
  if (i < n) dst[i] = f2bf(src[i]);
}

// ---------------------------------------------------------------------------
// Kernel 1: fused RMSNorm + sigmoid router + top-2 gather-list build
// One block (256 thr = 8 waves) per token.
// ---------------------------------------------------------------------------
__global__ __launch_bounds__(256)
void rms_router_kernel(const float* __restrict__ x,
                       const int*   __restrict__ is_visual,
                       const float* __restrict__ rms_w,
                       const float* __restrict__ router_w,   // (E,D)
                       const float* __restrict__ aux_bias,   // (E)
                       const float* __restrict__ mod_bias,   // (2,E)
                       bf16_t*      __restrict__ hbf,        // (N,D) out
                       int*         __restrict__ cnt,        // (E) pre-zeroed
                       int*         __restrict__ tokList,    // (E,N)
                       float*       __restrict__ gateList)   // (E,N)
{
  __shared__ __align__(16) float hsm[D_];
  __shared__ float red[8];
  __shared__ float logits[E_];

  const int n    = blockIdx.x;
  const int tid  = threadIdx.x;
  const int lane = tid & 31;
  const int wave = tid >> 5;
  const float* xr = x + (size_t)n * D_;

  float xv[4];
  float ss = 0.0f;
#pragma unroll
  for (int j = 0; j < 4; ++j) {
    xv[j] = xr[tid + 256 * j];
    ss += xv[j] * xv[j];
  }
#pragma unroll
  for (int o = 16; o > 0; o >>= 1) ss += __shfl_down(ss, o, 32);
  if (lane == 0) red[wave] = ss;
  __syncthreads();
  if (tid == 0) {
    float t = 0.0f;
#pragma unroll
    for (int w = 0; w < 8; ++w) t += red[w];
    red[0] = rsqrtf(t * (1.0f / (float)D_) + EPS_RMS_);
  }
  __syncthreads();
  const float rstd = red[0];

#pragma unroll
  for (int j = 0; j < 4; ++j) {
    int d = tid + 256 * j;
    float h = xv[j] * rstd * rms_w[d];
    hsm[d] = h;
    hbf[(size_t)n * D_ + d] = f2bf(h);
  }
  __syncthreads();

  // wave w -> router logit for expert w
  {
    const float* rw = router_w + (size_t)wave * D_;
    float acc = 0.0f;
    for (int d = lane; d < D_; d += 32) acc += hsm[d] * rw[d];
#pragma unroll
    for (int o = 16; o > 0; o >>= 1) acc += __shfl_down(acc, o, 32);
    if (lane == 0) logits[wave] = acc;
  }
  __syncthreads();

  if (tid == 0) {
    const int vis = is_visual[n];
    float aff[E_], biased[E_];
#pragma unroll
    for (int e = 0; e < E_; ++e) {
      float a = 1.0f / (1.0f + __expf(-logits[e]));
      aff[e]    = a;
      biased[e] = a + aux_bias[e] + mod_bias[vis * E_ + e];
    }
    int i0 = 0;
#pragma unroll
    for (int e = 1; e < E_; ++e) if (biased[e] > biased[i0]) i0 = e;
    int i1 = (i0 == 0) ? 1 : 0;
#pragma unroll
    for (int e = 0; e < E_; ++e) {
      if (e != i0 && biased[e] > biased[i1]) i1 = e;
    }
    float g0 = aff[i0], g1 = aff[i1];
    float inv = 1.0f / (g0 + g1 + 1e-12f);
    g0 *= inv; g1 *= inv;
    int p0 = atomicAdd(&cnt[i0], 1);
    tokList[i0 * N_TOK + p0]  = n;
    gateList[i0 * N_TOK + p0] = g0;
    int p1 = atomicAdd(&cnt[i1], 1);
    tokList[i1 * N_TOK + p1]  = n;
    gateList[i1 * N_TOK + p1] = g1;
  }
}

// ---------------------------------------------------------------------------
// Kernels 2/3: fused SwiGLU expert (shared dense / routed gathered)
// 32-token tile per workgroup, 256 threads = 8 waves, ~99 KB dynamic LDS.
// Stage 1: ACT(32xF) = silu(min(H Wg^T,30)) * clip(H Wu^T)  (WMMA -> LDS)
// Stage 2: OUT(32xD) = ACT Wd^T                             (WMMA)
// Both M-halves of the tile share one set of B (weight) fragments.
// ---------------------------------------------------------------------------
template <bool ROUTED>
__global__ __launch_bounds__(256)
void moe_expert_kernel(const bf16_t* __restrict__ hbf,     // (N,D) bf16
                      const bf16_t* __restrict__ wg_base, // (E,F,D) or (F,D)
                      const bf16_t* __restrict__ wu_base,
                      const bf16_t* __restrict__ wd_base, // (E,D,F) or (D,F)
                      const int*    __restrict__ cnt,
                      const int*    __restrict__ tokList,
                      const float*  __restrict__ gateList,
                      float*        __restrict__ out)     // (N,D) f32
{
  extern __shared__ __align__(16) char smem[];
  bf16_t* Hs     = reinterpret_cast<bf16_t*>(smem);
  bf16_t* As     = reinterpret_cast<bf16_t*>(smem + HS_BYTES);
  int*    s_tok  = reinterpret_cast<int*>(smem + HS_BYTES + ACT_BYTES);
  float*  s_gate = reinterpret_cast<float*>(smem + HS_BYTES + ACT_BYTES + MT_ * 4);

  const int tilesPer = N_TOK / MT_;  // 128
  int e = 0, tile;
  if (ROUTED) { e = blockIdx.x / tilesPer; tile = blockIdx.x % tilesPer; }
  else        { tile = blockIdx.x; }
  const int m0 = tile * MT_;

  int nvalid = MT_;
  if (ROUTED) {
    int c = cnt[e];
    if (m0 >= c) return;                 // block-uniform early exit
    nvalid = min(MT_, c - m0);
  }

  const size_t woff = ROUTED ? (size_t)e * F_ * D_ : 0;
  const bf16_t* Wg = wg_base + woff;
  const bf16_t* Wu = wu_base + woff;
  const bf16_t* Wd = wd_base + woff;    // D_*F_ == F_*D_ element stride

  const int tid = threadIdx.x;
  if (tid < MT_) {
    if (ROUTED) {
      bool v = tid < nvalid;
      s_tok[tid]  = v ? tokList[e * N_TOK + m0 + tid] : 0;
      s_gate[tid] = v ? gateList[e * N_TOK + m0 + tid] : 0.0f;
    } else {
      s_tok[tid]  = m0 + tid;
      s_gate[tid] = 1.0f;
    }
  }
  __syncthreads();

  // Stage gathered H rows into padded LDS via async DMA (ASYNCcnt path):
  // 8 threads per row, 16x16B chunks each.
  {
    const int r = tid >> 3;        // 0..31
    const int c = tid & 7;         // 0..7
    const bf16_t* src = hbf + (size_t)s_tok[r] * D_;
    bf16_t* dst = &Hs[r * HS_STRIDE];
#pragma unroll
    for (int j = 0; j < 16; ++j) {
      int elem = (c + 8 * j) * 8;  // 8 bf16 per 16B chunk
      async_g2l_b128(dst + elem, src + elem);
    }
  }
  wait_async_then_none();
  __syncthreads();

  const int wave = tid >> 5;
  const int lane = tid & 31;
  const int half = lane >> 4;
  const int lr   = lane & 15;

  // ---------------- Stage 1: G/U GEMMs + activation -> As ----------------
#pragma unroll
  for (int p = 0; p < 4; ++p) {
    const int nf = wave * 4 + p;        // 0..31 F-tiles
    const int f0 = nf * 16;
    const bf16_t* gRow = Wg + (size_t)(f0 + lr) * D_;
    const bf16_t* uRow = Wu + (size_t)(f0 + lr) * D_;
    const bf16_t* aRow0 = &Hs[lr * HS_STRIDE];
    const bf16_t* aRow1 = &Hs[(16 + lr) * HS_STRIDE];
    if (p < 3) {  // prefetch next F-tile's weight rows
      __builtin_prefetch(gRow + 16 * D_, 0, 0);
      __builtin_prefetch(uRow + 16 * D_, 0, 0);
    }
    v8f accG0 = {}, accU0 = {}, accG1 = {}, accU1 = {};
#pragma unroll 2
    for (int k0 = 0; k0 < D_; k0 += 32) {
      v16bf bg = frag_from(gRow, k0, half);
      v16bf bu = frag_from(uRow, k0, half);
      v16bf a0 = frag_from(aRow0, k0, half);
      v16bf a1 = frag_from(aRow1, k0, half);
      accG0 = __builtin_amdgcn_wmma_f32_16x16x32_bf16(false, a0, false, bg,
                                                      (short)0, accG0, false, false);
      accU0 = __builtin_amdgcn_wmma_f32_16x16x32_bf16(false, a0, false, bu,
                                                      (short)0, accU0, false, false);
      accG1 = __builtin_amdgcn_wmma_f32_16x16x32_bf16(false, a1, false, bg,
                                                      (short)0, accG1, false, false);
      accU1 = __builtin_amdgcn_wmma_f32_16x16x32_bf16(false, a1, false, bu,
                                                      (short)0, accU1, false, false);
    }
#pragma unroll
    for (int r = 0; r < 8; ++r) {
      const int m = r + 8 * half;       // C/D layout: vgpr r, lane half
      float g0 = fminf(accG0[r], GATE_MAX_);
      float u0 = fminf(fmaxf(accU0[r], LIN_MIN_), LIN_MAX_);
      As[m * ACT_STRIDE + f0 + lr] = f2bf((g0 / (1.0f + __expf(-g0))) * u0);
      float g1 = fminf(accG1[r], GATE_MAX_);
      float u1 = fminf(fmaxf(accU1[r], LIN_MIN_), LIN_MAX_);
      As[(m + 16) * ACT_STRIDE + f0 + lr] = f2bf((g1 / (1.0f + __expf(-g1))) * u1);
    }
  }
  __syncthreads();

  // ---------------- Stage 2: OUT = ACT @ Wd^T ----------------
#pragma unroll
  for (int q = 0; q < 8; ++q) {
    const int nd = wave * 8 + q;        // 0..63 D-tiles
    const int d0 = nd * 16;
    const bf16_t* dRow = Wd + (size_t)(d0 + lr) * F_;
    const bf16_t* aRow0 = &As[lr * ACT_STRIDE];
    const bf16_t* aRow1 = &As[(16 + lr) * ACT_STRIDE];
    if (q < 7) __builtin_prefetch(dRow + 16 * F_, 0, 0);
    v8f acc0 = {}, acc1 = {};
#pragma unroll 2
    for (int k0 = 0; k0 < F_; k0 += 32) {
      v16bf bd = frag_from(dRow, k0, half);
      v16bf a0 = frag_from(aRow0, k0, half);
      v16bf a1 = frag_from(aRow1, k0, half);
      acc0 = __builtin_amdgcn_wmma_f32_16x16x32_bf16(false, a0, false, bd,
                                                     (short)0, acc0, false, false);
      acc1 = __builtin_amdgcn_wmma_f32_16x16x32_bf16(false, a1, false, bd,
                                                     (short)0, acc1, false, false);
    }
#pragma unroll
    for (int r = 0; r < 8; ++r) {
      const int m = r + 8 * half;
      if (ROUTED) {
        unsafeAtomicAdd(&out[(size_t)s_tok[m] * D_ + d0 + lr], acc0[r] * s_gate[m]);
        unsafeAtomicAdd(&out[(size_t)s_tok[m + 16] * D_ + d0 + lr], acc1[r] * s_gate[m + 16]);
      } else {
        out[(size_t)s_tok[m] * D_ + d0 + lr]      = acc0[r];
        out[(size_t)s_tok[m + 16] * D_ + d0 + lr] = acc1[r];
      }
    }
  }
}

// ---------------------------------------------------------------------------
// Host launcher
// ---------------------------------------------------------------------------
extern "C" void kernel_launch(void* const* d_in, const int* in_sizes, int n_in,
                              void* d_out, int out_size, void* d_ws, size_t ws_size,
                              hipStream_t stream) {
  (void)in_sizes; (void)n_in; (void)out_size; (void)ws_size;

  const float* x         = (const float*)d_in[0];
  const int*   is_visual = (const int*)  d_in[1];
  const float* rms_w     = (const float*)d_in[2];
  const float* router_w  = (const float*)d_in[3];
  const float* aux_bias  = (const float*)d_in[4];
  const float* mod_bias  = (const float*)d_in[5];
  const float* sh_wg     = (const float*)d_in[6];
  const float* sh_wu     = (const float*)d_in[7];
  const float* sh_wd     = (const float*)d_in[8];
  const float* wg        = (const float*)d_in[9];
  const float* wu        = (const float*)d_in[10];
  const float* wd        = (const float*)d_in[11];
  float* out = (float*)d_out;

  char* ws = (char*)d_ws;
  size_t off = 0;
  auto alloc = [&](size_t bytes) -> void* {
    off = (off + 255) & ~(size_t)255;
    void* p = ws + off;
    off += bytes;
    return p;
  };
  bf16_t* hbf      = (bf16_t*)alloc((size_t)N_TOK * D_ * 2);
  bf16_t* shwg_bf  = (bf16_t*)alloc((size_t)F_ * D_ * 2);
  bf16_t* shwu_bf  = (bf16_t*)alloc((size_t)F_ * D_ * 2);
  bf16_t* shwd_bf  = (bf16_t*)alloc((size_t)D_ * F_ * 2);
  bf16_t* wg_bf    = (bf16_t*)alloc((size_t)E_ * F_ * D_ * 2);
  bf16_t* wu_bf    = (bf16_t*)alloc((size_t)E_ * F_ * D_ * 2);
  bf16_t* wd_bf    = (bf16_t*)alloc((size_t)E_ * D_ * F_ * 2);
  int*    cnt      = (int*)  alloc(E_ * sizeof(int));
  int*    tokList  = (int*)  alloc((size_t)E_ * N_TOK * sizeof(int));
  float*  gateList = (float*)alloc((size_t)E_ * N_TOK * sizeof(float));

  // Opt in to >64KB dynamic LDS (idempotent; not a stream op, capture-safe).
  hipFuncSetAttribute((const void*)moe_expert_kernel<false>,
                      hipFuncAttributeMaxDynamicSharedMemorySize, SMEM_TOTAL);
  hipFuncSetAttribute((const void*)moe_expert_kernel<true>,
                      hipFuncAttributeMaxDynamicSharedMemorySize, SMEM_TOTAL);

  hipMemsetAsync(cnt, 0, E_ * sizeof(int), stream);

  auto cvt = [&](const float* s, bf16_t* d, size_t ne) {
    cvt_bf16_kernel<<<(unsigned)((ne + 255) / 256), 256, 0, stream>>>(s, d, (int)ne);
  };
  cvt(sh_wg, shwg_bf, (size_t)F_ * D_);
  cvt(sh_wu, shwu_bf, (size_t)F_ * D_);
  cvt(sh_wd, shwd_bf, (size_t)D_ * F_);
  cvt(wg,    wg_bf,   (size_t)E_ * F_ * D_);
  cvt(wu,    wu_bf,   (size_t)E_ * F_ * D_);
  cvt(wd,    wd_bf,   (size_t)E_ * D_ * F_);

  rms_router_kernel<<<N_TOK, 256, 0, stream>>>(
      x, is_visual, rms_w, router_w, aux_bias, mod_bias,
      hbf, cnt, tokList, gateList);

  // Shared expert: dense over all tokens, plain stores (initializes d_out)
  moe_expert_kernel<false><<<N_TOK / MT_, 256, SMEM_TOTAL, stream>>>(
      hbf, shwg_bf, shwu_bf, shwd_bf, nullptr, nullptr, nullptr, out);

  // Routed experts: gathered tokens, gate-scaled atomic accumulation
  moe_expert_kernel<true><<<E_ * (N_TOK / MT_), 256, SMEM_TOTAL, stream>>>(
      hbf, wg_bf, wu_bf, wd_bf, cnt, tokList, gateList, out);
}
